// TorchMultiHeadAttention_5832565588132
// MI455X (gfx1250) — matrix operand order repaired and checked by reading, not compiled, not run on hardware
//
#include <hip/hip_runtime.h>

#define BB 2
#define SS 2048
#define DD 1024
#define HH 16
#define HD 64
#define MM (BB*SS)

typedef __attribute__((ext_vector_type(16))) __bf16    v16bf;
typedef __attribute__((ext_vector_type(8)))  __bf16    v8bf;
typedef __attribute__((ext_vector_type(8)))  float     v8f;
typedef __attribute__((ext_vector_type(4)))  unsigned  u32x4;
typedef __attribute__((ext_vector_type(2)))  unsigned  u32x2;
typedef __attribute__((ext_vector_type(4)))  float     f32x4;

#define DEV static __device__ __forceinline__

// fp32 -> bf16 RNE in the i32 domain (no bf16 scalars)
DEV unsigned rne_u32(float f){
  unsigned u = __float_as_uint(f);
  return u + (0x7FFFu + ((u >> 16) & 1u));
}
// pack two floats into one dword of 2 bf16 with a single v_perm_b32:
// result bytes {hi[3],hi[2],lo[3],lo[2]}  (selector 0x07060302, S0=hi, S1=lo)
DEV unsigned pk2(float lo, float hi){
  return __builtin_amdgcn_perm(rne_u32(hi), rne_u32(lo), 0x07060302u);
}
// scalar helpers for one-element stores / score rounding
DEV unsigned short f2bf(float f){ return (unsigned short)(rne_u32(f) >> 16); }
DEV float bf2f(unsigned short h){ return __uint_as_float(((unsigned)h) << 16); }

// 16-bit A-matrix 16x32 layout: lane holds row m=l&15;
// elems 0..7 = K[k0 + half*8 + 0..7], elems 8..15 = K[k0 + 16 + half*8 + 0..7]
// Built from two aligned 16B vector loads + concat (no scalar bf16 elements ever).
DEV v16bf load_a_frag(const unsigned short* row, int k0, int half){
  v8bf lo = *(const v8bf*)(row + k0 + half*8);
  v8bf hi = *(const v8bf*)(row + k0 + 16 + half*8);
  return __builtin_shufflevector(lo, hi, 0,1,2,3,4,5,6,7,8,9,10,11,12,13,14,15);
}
// 16-bit B-matrix 32x16 layout: lane holds col n=l&15; same K order as A-frag:
// elems 0..7 = K[k0 + half*8 + 0..7], elems 8..15 = K[k0 + 16 + half*8 + 0..7]; source row n contiguous in k (B[k][n]=src[n][k])
DEV v16bf load_b_frag(const unsigned short* row, int k0, int half){
  v8bf lo = *(const v8bf*)(row + k0 + half*8);
  v8bf hi = *(const v8bf*)(row + k0 + 16 + half*8);
  return __builtin_shufflevector(lo, hi, 0,1,2,3,4,5,6,7,8,9,10,11,12,13,14,15);
}
DEV v8f wmma_bf16(v16bf a, v16bf b, v8f c){
  return __builtin_amdgcn_wmma_f32_16x16x32_bf16(false, a, false, b, (short)0, c,
                                                 false, false);
}

// ---------------- fp32 -> bf16 staging (pure i32 packing, dword stores) ----------
__global__ __launch_bounds__(256)
void cvt_bf16_kernel(const float* __restrict__ src, unsigned* __restrict__ dst, int n4){
  int i = blockIdx.x*256 + threadIdx.x;
  if(i >= n4) return;
  f32x4 v = *(const f32x4*)(src + (size_t)i*4);
  u32x2 w;
  w[0] = pk2(v[0], v[1]);
  w[1] = pk2(v[2], v[3]);
  *(u32x2*)(dst + (size_t)i*2) = w;
}

// ---------------- projection GEMM: out = A @ W.T + bias, bf16 out ---------------
// vmode 0: head-split dst[(b*HH+h)*SS + l][d]   (Q, K)
// vmode 1: per-head transposed dst[(b*HH+h)*HD + d][l]   (V, for contiguous P@V B-frags)
__global__ __launch_bounds__(32)
void proj_gemm_kernel(const unsigned short* __restrict__ A,
                      const unsigned short* __restrict__ W,
                      const float* __restrict__ bias, unsigned short* __restrict__ dst,
                      int vmode){
  const int lane = threadIdx.x, l15 = lane & 15, half = lane >> 4;
  const int m0 = blockIdx.y * 64, n0 = blockIdx.x * 64;
  v8f acc[4][4];
  for(int i=0;i<4;i++)for(int j=0;j<4;j++)for(int r=0;r<8;r++) acc[i][j][r]=0.f;
  const unsigned short* arow[4]; const unsigned short* brow[4];
  for(int i=0;i<4;i++) arow[i] = A + (size_t)(m0 + i*16 + l15)*DD;
  for(int j=0;j<4;j++) brow[j] = W + (size_t)(n0 + j*16 + l15)*DD;
  for(int k0=0;k0<DD;k0+=32){
    v16bf af[4], bfr[4];
    for(int i=0;i<4;i++) af[i]  = load_a_frag(arow[i], k0, half);
    for(int j=0;j<4;j++) bfr[j] = load_b_frag(brow[j], k0, half);
    for(int i=0;i<4;i++)
      for(int j=0;j<4;j++)
        acc[i][j] = wmma_bf16(af[i], bfr[j], acc[i][j]);
  }
  for(int j=0;j<4;j++){
    int n = n0 + j*16 + l15;
    int h = n >> 6, dcol = n & (HD-1);
    float bv = bias[n];
    for(int i=0;i<4;i++){
      int mbase = m0 + i*16 + half*8;
      int b = mbase >> 11, l = mbase & (SS-1);
      if(vmode == 0){
        size_t off = ((size_t)(b*HH + h)*SS + l)*HD + dcol;
        for(int r=0;r<8;r++)
          dst[off + (size_t)r*HD] = f2bf(acc[i][j][r] + bv);
      } else {
        size_t off = ((size_t)(b*HH + h)*HD + dcol)*SS + l; // 8 consecutive l -> b128 store
        u32x4 pkv;
        pkv[0] = pk2(acc[i][j][0]+bv, acc[i][j][1]+bv);
        pkv[1] = pk2(acc[i][j][2]+bv, acc[i][j][3]+bv);
        pkv[2] = pk2(acc[i][j][4]+bv, acc[i][j][5]+bv);
        pkv[3] = pk2(acc[i][j][6]+bv, acc[i][j][7]+bv);
        *(u32x4*)(dst + off) = pkv;
      }
    }
  }
}

// ---------------- output GEMM: out_f32 = ctx_bf16 @ Wo_bf16.T + bo ---------------
__global__ __launch_bounds__(32)
void out_gemm_kernel(const unsigned short* __restrict__ A,
                     const unsigned short* __restrict__ W,
                     const float* __restrict__ bias, float* __restrict__ out){
  const int lane = threadIdx.x, l15 = lane & 15, half = lane >> 4;
  const int m0 = blockIdx.y * 64, n0 = blockIdx.x * 64;
  v8f acc[4][4];
  for(int i=0;i<4;i++)for(int j=0;j<4;j++)for(int r=0;r<8;r++) acc[i][j][r]=0.f;
  const unsigned short* arow[4]; const unsigned short* brow[4];
  for(int i=0;i<4;i++) arow[i] = A + (size_t)(m0 + i*16 + l15)*DD;
  for(int j=0;j<4;j++) brow[j] = W + (size_t)(n0 + j*16 + l15)*DD;
  for(int k0=0;k0<DD;k0+=32){
    v16bf af[4], bfr[4];
    for(int i=0;i<4;i++) af[i]  = load_a_frag(arow[i], k0, half);
    for(int j=0;j<4;j++) bfr[j] = load_b_frag(brow[j], k0, half);
    for(int i=0;i<4;i++)
      for(int j=0;j<4;j++)
        acc[i][j] = wmma_bf16(af[i], bfr[j], acc[i][j]);
  }
  for(int j=0;j<4;j++){
    int n = n0 + j*16 + l15;
    float bv = bias[n];
    for(int i=0;i<4;i++){
      int mbase = m0 + i*16 + half*8;
      for(int r=0;r<8;r++)
        out[(size_t)(mbase + r)*DD + n] = acc[i][j][r] + bv;
    }
  }
}

// ---------------- q-norm over head dim ------------------------------------------
__global__ void qnorm_kernel(const unsigned short* __restrict__ Qh, float* __restrict__ qn){
  int i = blockIdx.x*blockDim.x + threadIdx.x;   // over BB*HH*SS
  const unsigned short* p = Qh + (size_t)i*HD;
  float s = 0.f;
  for(int d=0; d<HD; d+=8){
    u32x4 w = *(const u32x4*)(p + d);
    for(int t=0;t<4;t++){
      float a = __uint_as_float(w[t] << 16);
      float b = __uint_as_float(w[t] & 0xFFFF0000u);
      s += a*a + b*b;
    }
  }
  qn[i] = s;
}

// ---------------- ProbSparse top-u mask via stable rank counting -----------------
__global__ __launch_bounds__(256)
void psmask_kernel(const float* __restrict__ qn, const int* __restrict__ qvl,
                   float* __restrict__ msk){
  __shared__ float row[SS];
  const int bh = blockIdx.x;
  const float* q = qn + (size_t)bh*SS;
  for(int i=threadIdx.x;i<SS;i+=256) row[i]=q[i];
  __syncthreads();
  int qv = qvl[0]; if(qv < 2) qv = 2;
  int u = (int)(5.0f * logf((float)qv));          // FACTOR * log(q_valid_len)
  if(u < 1) u = 1; if(u > SS) u = SS;
  for(int i=threadIdx.x;i<SS;i+=256){
    float vi = row[i]; int cnt = 0;
    for(int j=0;j<SS;j++){
      float vj = row[j];
      cnt += (vj > vi) || (vj == vi && j < i);    // stable top_k tie-break
    }
    msk[(size_t)bh*SS + i] = (cnt < u) ? 1.f : 0.f;
  }
}

// ---------------- flash attention: one wave per (b,h, 32-query tile) -------------
#define ATT_STR 40   // LDS row stride (bf16 elems): keeps 16B alignment, spreads banks
__global__ __launch_bounds__(32)
void attn_kernel(const unsigned short* __restrict__ Qh,
                 const unsigned short* __restrict__ Kh,
                 const unsigned short* __restrict__ Vt,
                 const float* __restrict__ maskin,   // [S,S]
                 const float* __restrict__ psmask,   // [B*H*S]
                 const int* __restrict__ kvlp,
                 unsigned short* __restrict__ ctx){  // [B,S,D] merged heads, bf16
  __shared__ unsigned short pl[32*ATT_STR];
  const int lane = threadIdx.x, l15 = lane & 15, half = lane >> 4;
  const int q0 = blockIdx.x * 32;
  const int h = blockIdx.y, b = blockIdx.z;
  const int bh = b*HH + h;
  const int kvl = kvlp[0];
  const unsigned short* Qb = Qh + (size_t)bh*SS*HD;
  const unsigned short* Kb = Kh + (size_t)bh*SS*HD;
  const unsigned short* Vb = Vt + (size_t)bh*HD*SS;

  v16bf qa[2][2];
  for(int i=0;i<2;i++){
    const unsigned short* r = Qb + (size_t)(q0 + i*16 + l15)*HD;
    for(int t=0;t<2;t++) qa[i][t] = load_a_frag(r, t*32, half);
  }
  v8f o[2][4]; float mrow[2][8], lrow[2][8];
  for(int i=0;i<2;i++){
    for(int j=0;j<4;j++) for(int r=0;r<8;r++) o[i][j][r]=0.f;
    for(int r=0;r<8;r++){ mrow[i][r]=-3.0e38f; lrow[i][r]=0.f; }
  }

  for(int kt=0; kt<SS; kt+=32){
    // ---- scores S = Q @ K^T over this 32-key slab (8 WMMAs) ----
    v8f s[2][2];
    for(int i=0;i<2;i++)for(int jk=0;jk<2;jk++)for(int r=0;r<8;r++) s[i][jk][r]=0.f;
    for(int t=0;t<2;t++){
      v16bf kb0 = load_b_frag(Kb + (size_t)(kt +      l15)*HD, t*32, half);
      v16bf kb1 = load_b_frag(Kb + (size_t)(kt + 16 + l15)*HD, t*32, half);
      for(int i=0;i<2;i++){
        s[i][0] = wmma_bf16(qa[i][t], kb0, s[i][0]);
        s[i][1] = wmma_bf16(qa[i][t], kb1, s[i][1]);
      }
    }
    if (kt + 32 < SS)  // pull next K slab toward the caches
      __builtin_prefetch(Kb + (size_t)(kt + 32 + l15)*HD, 0, 0);

    // ---- scale, bf16-round (ref casts scores to bf16), mask, online softmax ----
    float p[2][2][8], corr[2][8];
    for(int i=0;i<2;i++){
      float rmax[8];
      for(int r=0;r<8;r++){
        int qrow = q0 + i*16 + half*8 + r;
        float best = -3.0e38f;
        for(int jk=0;jk<2;jk++){
          int kcol = kt + jk*16 + l15;
          float mv = maskin[(size_t)qrow*SS + kcol] + ((kcol >= kvl) ? -1e9f : 0.f);
          float sv = bf2f(f2bf(s[i][jk][r] * 0.125f));          // /sqrt(64), bf16 round
          sv = bf2f(f2bf(sv + bf2f(f2bf(mv))));                 // bf16 add of bf16 mask
          s[i][jk][r] = sv;
          best = fmaxf(best, sv);
        }
        rmax[r] = best;
      }
      for(int r=0;r<8;r++){                                     // 16-lane row reductions
        float v = rmax[r];
        v = fmaxf(v, __shfl_xor(v,1,32));
        v = fmaxf(v, __shfl_xor(v,2,32));
        v = fmaxf(v, __shfl_xor(v,4,32));
        v = fmaxf(v, __shfl_xor(v,8,32));
        rmax[r] = v;
      }
      for(int r=0;r<8;r++){
        float mnew = fmaxf(mrow[i][r], rmax[r]);
        float c = __expf(mrow[i][r] - mnew);
        corr[i][r] = c; mrow[i][r] = mnew;
        float psum = 0.f;
        for(int jk=0;jk<2;jk++){
          float e = __expf(s[i][jk][r] - mnew);
          p[i][jk][r] = e; psum += e;
        }
        psum += __shfl_xor(psum,1,32);
        psum += __shfl_xor(psum,2,32);
        psum += __shfl_xor(psum,4,32);
        psum += __shfl_xor(psum,8,32);
        lrow[i][r] = lrow[i][r]*c + psum;
      }
      for(int j=0;j<4;j++) for(int r=0;r<8;r++) o[i][j][r] *= corr[i][r];
    }

    // ---- C-layout -> A-layout relayout of P through LDS ----
    for(int i=0;i<2;i++)for(int jk=0;jk<2;jk++)for(int r=0;r<8;r++){
      int row = i*16 + half*8 + r;
      int col = jk*16 + l15;
      pl[row*ATT_STR + col] = f2bf(p[i][jk][r]);
    }
    __syncthreads();
    v16bf pa[2][2];
    for(int i=0;i<2;i++){
      const unsigned short* pr = pl + (i*16 + l15)*ATT_STR;
      for(int t=0;t<2;t++) pa[i][t] = load_a_frag(pr, t*32, half);
    }
    __syncthreads();

    // ---- O += P @ V  (V pre-transposed -> contiguous B-frag loads; 16 WMMAs) ----
    for(int t=0;t<2;t++){
      for(int j=0;j<4;j++){
        v16bf vbf = load_b_frag(Vb + (size_t)(j*16 + l15)*SS, kt + t*32, half);
        for(int i=0;i<2;i++) o[i][j] = wmma_bf16(pa[i][t], vbf, o[i][j]);
      }
    }
  }

  // ---- normalize, apply ProbSparse mask, store merged-head context (bf16) ----
  for(int i=0;i<2;i++){
    float inv[8];
    for(int r=0;r<8;r++){
      int qrow = q0 + i*16 + half*8 + r;
      inv[r] = psmask[(size_t)bh*SS + qrow] / lrow[i][r];
    }
    for(int j=0;j<4;j++){
      int col = h*HD + j*16 + l15;
      for(int r=0;r<8;r++){
        int qrow = q0 + i*16 + half*8 + r;
        ctx[((size_t)(b*SS + qrow))*DD + col] = f2bf(o[i][j][r]*inv[r]);
      }
    }
  }
}

extern "C" void kernel_launch(void* const* d_in, const int* in_sizes, int n_in,
                              void* d_out, int out_size, void* d_ws, size_t ws_size,
                              hipStream_t stream) {
  (void)in_sizes; (void)n_in; (void)out_size; (void)ws_size;
  const float* query = (const float*)d_in[0];
  const float* key_  = (const float*)d_in[1];
  const float* value = (const float*)d_in[2];
  const float* mask  = (const float*)d_in[3];
  const float* Wq = (const float*)d_in[4];  const float* bq = (const float*)d_in[5];
  const float* Wk = (const float*)d_in[6];  const float* bk = (const float*)d_in[7];
  const float* Wv = (const float*)d_in[8];  const float* bv = (const float*)d_in[9];
  const float* Wo = (const float*)d_in[10]; const float* bo = (const float*)d_in[11];
  const int* qvl = (const int*)d_in[12];
  const int* kvl = (const int*)d_in[13];

  char* ws = (char*)d_ws;
  const size_t MB = 1024ull*1024ull;
  unsigned short* qbf = (unsigned short*)(ws + 0*MB);   // [B*S,D] bf16, 8MB
  unsigned short* kbf = (unsigned short*)(ws + 8*MB);   // 8MB
  unsigned short* vbf = (unsigned short*)(ws + 16*MB);  // 8MB
  unsigned short* wqb = (unsigned short*)(ws + 24*MB);  // [D,D] bf16, 2MB
  unsigned short* wkb = (unsigned short*)(ws + 26*MB);
  unsigned short* wvb = (unsigned short*)(ws + 28*MB);
  unsigned short* wob = (unsigned short*)(ws + 30*MB);
  unsigned short* Qh  = (unsigned short*)(ws + 32*MB);  // [B,H,S,64] bf16, 8MB
  unsigned short* Kh  = (unsigned short*)(ws + 40*MB);  // 8MB
  unsigned short* Vt  = (unsigned short*)(ws + 48*MB);  // [B,H,64,S] bf16, 8MB
  unsigned short* ctx = (unsigned short*)(ws + 56*MB);  // [B,S,D] bf16, 8MB
  float* qn  = (float*)(ws + 64*MB);                    // [B*H*S] f32, 256KB
  float* pmk = (float*)(ws + 64*MB + 256*1024);         // [B*H*S] f32, 256KB

  // stage activations + weights as bf16 (one bandwidth pass, ~64MB total)
  const int nact4 = MM*DD/4, nw4 = DD*DD/4;
  cvt_bf16_kernel<<<nact4/256, 256, 0, stream>>>(query, (unsigned*)qbf, nact4);
  cvt_bf16_kernel<<<nact4/256, 256, 0, stream>>>(key_,  (unsigned*)kbf, nact4);
  cvt_bf16_kernel<<<nact4/256, 256, 0, stream>>>(value, (unsigned*)vbf, nact4);
  cvt_bf16_kernel<<<nw4/256,  256, 0, stream>>>(Wq, (unsigned*)wqb, nw4);
  cvt_bf16_kernel<<<nw4/256,  256, 0, stream>>>(Wk, (unsigned*)wkb, nw4);
  cvt_bf16_kernel<<<nw4/256,  256, 0, stream>>>(Wv, (unsigned*)wvb, nw4);
  cvt_bf16_kernel<<<nw4/256,  256, 0, stream>>>(Wo, (unsigned*)wob, nw4);

  dim3 g1(DD/64, MM/64);                                // 16 x 64 tiles, 1 wave each
  proj_gemm_kernel<<<g1, 32, 0, stream>>>(qbf, wqb, bq, Qh, 0);
  proj_gemm_kernel<<<g1, 32, 0, stream>>>(kbf, wkb, bk, Kh, 0);
  proj_gemm_kernel<<<g1, 32, 0, stream>>>(vbf, wvb, bv, Vt, 1);

  qnorm_kernel<<<(BB*HH*SS)/256, 256, 0, stream>>>(Qh, qn);
  psmask_kernel<<<BB*HH, 256, 0, stream>>>(qn, qvl, pmk);

  dim3 ga(SS/32, HH, BB);                               // 2048 waves
  attn_kernel<<<ga, 32, 0, stream>>>(Qh, Kh, Vt, mask, pmk, kvl, ctx);

  out_gemm_kernel<<<g1, 32, 0, stream>>>(ctx, wob, bo, (float*)d_out);
}
